// CGATFusion_52055003627699
// MI455X (gfx1250) — matrix operand order repaired and checked by reading, not compiled, not run on hardware
//
#include <hip/hip_runtime.h>
#include <hip/hip_bf16.h>

typedef _Float16 v16h __attribute__((ext_vector_type(16)));
typedef float    v8f  __attribute__((ext_vector_type(8)));

#define NS_    2048
#define NC_    4096
#define NG_    2048
#define DM_    256
#define NH_    8
#define DK_    32
#define TT_    4097
#define NSLAB_ 8
#define CSLAB_ (NC_ / NSLAB_)   // 512

static __device__ __forceinline__ v8f wmma_f16(v16h a, v16h b, v8f c) {
  return __builtin_amdgcn_wmma_f32_16x16x32_f16(false, a, false, b, (short)0, c, false, false);
}

// ---------------------------------------------------------------------------
// Generic Y = X @ W^T + bias   (X: MxK f32, W: NxK f32, Y: MxN f32)
// One wave per 16x16 output tile, K consumed 32 at a time via f16 WMMA.
// ---------------------------------------------------------------------------
__global__ void gemm_xwT_bias(const float* __restrict__ X, const float* __restrict__ W,
                              const float* __restrict__ bias, float* __restrict__ Y,
                              int M, int K, int N) {
  const int lane = threadIdx.x;
  const int l15 = lane & 15, g = lane >> 4;
  const int ntile = blockIdx.x * 16;
  const int mtile = blockIdx.y * 16;
  const int arow = mtile + l15;   // A-layout: this lane's row
  const int bcol = ntile + l15;   // B-layout: this lane's column
  const bool avalid = (arow < M);
  v8f acc = {};
  for (int k0 = 0; k0 < K; k0 += 32) {
    v16h a, b;
    if (avalid) {
      const float* xp = X + (size_t)arow * K + k0 + g * 8;
      __builtin_prefetch((const void*)(xp + 32), 0, 1);   // next K chunk
#pragma unroll
      for (int i = 0; i < 8; ++i) { a[i] = (_Float16)xp[i]; a[8 + i] = (_Float16)xp[16 + i]; }
    } else {
#pragma unroll
      for (int i = 0; i < 16; ++i) a[i] = (_Float16)0.0f;
    }
    const float* wp = W + (size_t)bcol * K + k0 + g * 8;
    __builtin_prefetch((const void*)(wp + 32), 0, 1);
#pragma unroll
    for (int i = 0; i < 8; ++i) { b[i] = (_Float16)wp[i]; b[8 + i] = (_Float16)wp[16 + i]; }
    acc = wmma_f16(a, b, acc);
  }
  const float bv = bias ? bias[bcol] : 0.0f;
#pragma unroll
  for (int i = 0; i < 8; ++i) {
    const int row = mtile + g * 8 + i;
    if (row < M) Y[(size_t)row * N + bcol] = acc[i] + bv;
  }
}

// ---------------------------------------------------------------------------
// Cross-attention pass 1: online max / sumexp of gated scores per (s, head).
// score(s,c) = (Q[s,h].K[c,h] / sqrt(DK)) * G[s,c]   (gate BEFORE softmax)
// One wave per (16-row s-tile, head); one WMMA per 16-column chunk of N_C.
// ---------------------------------------------------------------------------
__global__ void xattn_stats(const float* __restrict__ Qm, const float* __restrict__ Km,
                            const float* __restrict__ G, float* __restrict__ rowmax,
                            float* __restrict__ rowinv) {
  const int lane = threadIdx.x;
  const int l15 = lane & 15, g = lane >> 4;
  const int stile = blockIdx.x * 16;
  const int h = blockIdx.y;
  const float scale = 0.17677669529663687f;   // 1/sqrt(32), folded into Q
  v16h qa;
  {
    const float* qp = Qm + (size_t)(stile + l15) * DM_ + h * DK_ + g * 8;
#pragma unroll
    for (int i = 0; i < 8; ++i) {
      qa[i]     = (_Float16)(qp[i] * scale);
      qa[8 + i] = (_Float16)(qp[16 + i] * scale);
    }
  }
  float rm[8], rs[8];
#pragma unroll
  for (int i = 0; i < 8; ++i) { rm[i] = -1e30f; rs[i] = 0.0f; }

  for (int c0 = 0; c0 < NC_; c0 += 16) {
    v16h kb;
    const float* kp = Km + (size_t)(c0 + l15) * DM_ + h * DK_ + g * 8;
    __builtin_prefetch((const void*)(kp + 16 * DM_), 0, 1);   // next 16 K-rows
#pragma unroll
    for (int i = 0; i < 8; ++i) { kb[i] = (_Float16)kp[i]; kb[8 + i] = (_Float16)kp[16 + i]; }
    v8f s = {};
    s = wmma_f16(qa, kb, s);
    const float* gp = G + (size_t)(stile + g * 8) * NC_ + c0 + l15;
#pragma unroll
    for (int i = 0; i < 8; ++i) {
      const float sv = s[i] * gp[(size_t)i * NC_];
      float cm = sv;
      for (int m = 1; m < 16; m <<= 1) cm = fmaxf(cm, __shfl_xor(cm, m, 32));
      const float nm = fmaxf(rm[i], cm);
      float e = __expf(sv - nm);
      for (int m = 1; m < 16; m <<= 1) e += __shfl_xor(e, m, 32);
      rs[i] = rs[i] * __expf(rm[i] - nm) + e;
      rm[i] = nm;
    }
  }
  if (l15 == 0) {
#pragma unroll
    for (int i = 0; i < 8; ++i) {
      const int row = stile + g * 8 + i;
      rowmax[row * NH_ + h] = rm[i];
      rowinv[row * NH_ + h] = 1.0f / rs[i];
    }
  }
}

// ---------------------------------------------------------------------------
// Cross-attention pass 2: recompute scores, normalize, emit attn_map
// (mean over heads) and per-slab partials of H_new = attn @ V.
// Grid: (s-tiles, NSLAB_ column slabs) -> 1024 waves; each block exclusively
// owns a 16 x 512 region of attn_map (heads accumulated in-block, no atomics).
// ---------------------------------------------------------------------------
__global__ void xattn_apply(const float* __restrict__ Qm, const float* __restrict__ Km,
                            const float* __restrict__ Vm, const float* __restrict__ G,
                            const float* __restrict__ rowmax, const float* __restrict__ rowinv,
                            float* __restrict__ amap, float* __restrict__ Hpart) {
  __shared__ _Float16 atile[16 * 32];
  __shared__ float smax[16 * NH_];
  __shared__ float sinv[16 * NH_];
  const int lane = threadIdx.x;
  const int l15 = lane & 15, g = lane >> 4;
  const int stile = blockIdx.x * 16;
  const int slab  = blockIdx.y;
  const int cbeg  = slab * CSLAB_;
  const float scale = 0.17677669529663687f;
  for (int idx = lane; idx < 16 * NH_; idx += 32) {
    smax[idx] = rowmax[stile * NH_ + idx];
    sinv[idx] = rowinv[stile * NH_ + idx];
  }
  __syncthreads();

  float* hp = Hpart + (size_t)slab * ((size_t)NS_ * DM_);

  for (int h = 0; h < NH_; ++h) {
    v16h qa;
    {
      const float* qp = Qm + (size_t)(stile + l15) * DM_ + h * DK_ + g * 8;
#pragma unroll
      for (int i = 0; i < 8; ++i) {
        qa[i]     = (_Float16)(qp[i] * scale);
        qa[8 + i] = (_Float16)(qp[16 + i] * scale);
      }
    }
    v8f acc0 = {}, acc1 = {};
    for (int c0 = cbeg; c0 < cbeg + CSLAB_; c0 += 32) {
      // two 16x16 score tiles (columns c0..c0+15 and c0+16..c0+31)
      v16h kb0, kb1;
      const float* kp0 = Km + (size_t)(c0 + l15) * DM_ + h * DK_ + g * 8;
      const float* kp1 = kp0 + (size_t)16 * DM_;
      __builtin_prefetch((const void*)(kp0 + 32 * DM_), 0, 1);
#pragma unroll
      for (int i = 0; i < 8; ++i) { kb0[i] = (_Float16)kp0[i]; kb1[i] = (_Float16)kp1[i]; kb0[8 + i] = (_Float16)kp0[16 + i]; kb1[8 + i] = (_Float16)kp1[16 + i]; }
      v8f s0 = {}, s1 = {};
      s0 = wmma_f16(qa, kb0, s0);
      s1 = wmma_f16(qa, kb1, s1);

      const float* gp = G + (size_t)(stile + g * 8) * NC_ + c0 + l15;
      float* op = amap + (size_t)(stile + g * 8) * NC_ + c0 + l15;
#pragma unroll
      for (int i = 0; i < 8; ++i) {
        const int ridx = (g * 8 + i) * NH_ + h;
        const float mh = smax[ridx], iv = sinv[ridx];
        const float a0 = __expf(s0[i] * gp[(size_t)i * NC_]      - mh) * iv;
        const float a1 = __expf(s1[i] * gp[(size_t)i * NC_ + 16] - mh) * iv;
        if (h == 0) {
          op[(size_t)i * NC_]      = 0.125f * a0;
          op[(size_t)i * NC_ + 16] = 0.125f * a1;
        } else {
          op[(size_t)i * NC_]      += 0.125f * a0;
          op[(size_t)i * NC_ + 16] += 0.125f * a1;
        }
        atile[(g * 8 + i) * 32 + l15]      = (_Float16)a0;
        atile[(g * 8 + i) * 32 + 16 + l15] = (_Float16)a1;
      }
      __syncthreads();
      // reload attn in A-layout, multiply by V chunk (32 c x 32 d)
      v16h at;
      const _Float16* ap = &atile[l15 * 32 + g * 8];
#pragma unroll
      for (int i = 0; i < 8; ++i) { at[i] = ap[i]; at[8 + i] = ap[16 + i]; }
      v16h vb0, vb1;
      const float* vp = Vm + (size_t)(c0 + g * 8) * DM_ + h * DK_ + l15;
      __builtin_prefetch((const void*)(vp + 32 * DM_), 0, 1);
#pragma unroll
      for (int i = 0; i < 16; ++i) {
        const int kr = (i < 8) ? i : (i + 8);
        vb0[i] = (_Float16)vp[(size_t)kr * DM_];
        vb1[i] = (_Float16)vp[(size_t)kr * DM_ + 16];
      }
      acc0 = wmma_f16(at, vb0, acc0);
      acc1 = wmma_f16(at, vb1, acc1);
      __syncthreads();
    }
#pragma unroll
    for (int i = 0; i < 8; ++i) {
      const int row = stile + g * 8 + i;
      hp[(size_t)row * DM_ + h * DK_ + l15]      = acc0[i];
      hp[(size_t)row * DM_ + h * DK_ + 16 + l15] = acc1[i];
    }
  }
}

// Deterministic cross-slab reduction: Hnew = sum_p Hpart[p]
__global__ void hnew_reduce(const float* __restrict__ Hpart, float* __restrict__ Hnew) {
  const size_t idx = (size_t)blockIdx.x * blockDim.x + threadIdx.x;
  float s = 0.0f;
#pragma unroll
  for (int p = 0; p < NSLAB_; ++p) s += Hpart[(size_t)p * ((size_t)NS_ * DM_) + idx];
  Hnew[idx] = s;
}

// ---------------------------------------------------------------------------
// Row LayerNorm of (x + res) with gain/bias. One 256-thread block per row.
// ---------------------------------------------------------------------------
__global__ void ln_res_kernel(const float* __restrict__ x, const float* __restrict__ res,
                              const float* __restrict__ gw, const float* __restrict__ bw,
                              float* __restrict__ y) {
  __shared__ float red[DM_];
  const int row = blockIdx.x, t = threadIdx.x;
  float v = x[(size_t)row * DM_ + t];
  if (res) v += res[(size_t)row * DM_ + t];
  red[t] = v; __syncthreads();
  for (int s = DM_ / 2; s > 0; s >>= 1) { if (t < s) red[t] += red[t + s]; __syncthreads(); }
  const float mean = red[0] * (1.0f / DM_);
  __syncthreads();
  const float d = v - mean;
  red[t] = d * d; __syncthreads();
  for (int s = DM_ / 2; s > 0; s >>= 1) { if (t < s) red[t] += red[t + s]; __syncthreads(); }
  const float var = red[0] * (1.0f / DM_);
  y[(size_t)row * DM_ + t] = d * rsqrtf(var + 1e-5f) * gw[t] + bw[t];
}

// tokens = concat(fusion_token, HSu, HG)
__global__ void build_tokens(const float* __restrict__ fus, const float* __restrict__ HSu,
                             const float* __restrict__ HG, float* __restrict__ tokens) {
  const int idx = blockIdx.x * blockDim.x + threadIdx.x;
  if (idx >= TT_ * DM_) return;
  const int row = idx / DM_, col = idx - row * DM_;
  float v;
  if (row == 0)          v = fus[col];
  else if (row <= NS_)   v = HSu[(size_t)(row - 1) * DM_ + col];
  else                   v = HG[(size_t)(row - 1 - NS_) * DM_ + col];
  tokens[idx] = v;
}

// ---------------------------------------------------------------------------
// Final head: token-0 self-attention over all 4097 keys, out-proj, LN1,
// FFN(gelu exact), LN2 -> z_fused[256]. Single 256-thread block (tiny work).
// ---------------------------------------------------------------------------
__global__ void fusion_head_kernel(const float* __restrict__ qkv, const float* __restrict__ tokens,
                                   const float* __restrict__ mow, const float* __restrict__ mob,
                                   const float* __restrict__ ln1g, const float* __restrict__ ln1b,
                                   const float* __restrict__ w1, const float* __restrict__ b1,
                                   const float* __restrict__ w2, const float* __restrict__ b2,
                                   const float* __restrict__ ln2g, const float* __restrict__ ln2b,
                                   float* __restrict__ z) {
  __shared__ float q0[DM_];
  __shared__ float red[DM_];
  __shared__ float hmax[NH_], hsum[NH_];
  __shared__ float attn_out[DM_];
  __shared__ float t1[DM_];
  __shared__ float hbuf[2 * DM_];
  const float scale = 0.17677669529663687f;
  const int t = threadIdx.x;
  q0[t] = qkv[t];
  __syncthreads();

  float lmax[NH_], lsum[NH_];
#pragma unroll
  for (int h = 0; h < NH_; ++h) { lmax[h] = -1e30f; lsum[h] = 0.0f; }
  for (int u = t; u < TT_; u += DM_) {
    const float* kp = qkv + (size_t)u * (3 * DM_) + DM_;
#pragma unroll
    for (int h = 0; h < NH_; ++h) {
      float d = 0.0f;
      for (int j = 0; j < DK_; ++j) d += q0[h * DK_ + j] * kp[h * DK_ + j];
      lmax[h] = fmaxf(lmax[h], d * scale);
    }
  }
  for (int h = 0; h < NH_; ++h) {
    red[t] = lmax[h]; __syncthreads();
    for (int s = DM_ / 2; s > 0; s >>= 1) { if (t < s) red[t] = fmaxf(red[t], red[t + s]); __syncthreads(); }
    if (t == 0) hmax[h] = red[0];
    __syncthreads();
  }
  for (int u = t; u < TT_; u += DM_) {
    const float* kp = qkv + (size_t)u * (3 * DM_) + DM_;
#pragma unroll
    for (int h = 0; h < NH_; ++h) {
      float d = 0.0f;
      for (int j = 0; j < DK_; ++j) d += q0[h * DK_ + j] * kp[h * DK_ + j];
      lsum[h] += __expf(d * scale - hmax[h]);
    }
  }
  for (int h = 0; h < NH_; ++h) {
    red[t] = lsum[h]; __syncthreads();
    for (int s = DM_ / 2; s > 0; s >>= 1) { if (t < s) red[t] += red[t + s]; __syncthreads(); }
    if (t == 0) hsum[h] = red[0];
    __syncthreads();
  }
  {
    const int h = t >> 5, dk = t & 31;
    const float m = hmax[h], inv = 1.0f / hsum[h];
    float acc = 0.0f;
    for (int u = 0; u < TT_; ++u) {
      const float* kp = qkv + (size_t)u * (3 * DM_) + DM_ + h * DK_;
      float d = 0.0f;
      for (int j = 0; j < DK_; ++j) d += q0[h * DK_ + j] * kp[j];
      acc += __expf(d * scale - m) * kp[DM_ + dk];   // v = k + 256 within the row
    }
    attn_out[t] = acc * inv;
  }
  __syncthreads();
  // out projection + residual
  float y = mob[t];
  for (int d = 0; d < DM_; ++d) y += attn_out[d] * mow[(size_t)t * DM_ + d];
  const float v0 = tokens[t] + y;
  // LN1
  red[t] = v0; __syncthreads();
  for (int s = DM_ / 2; s > 0; s >>= 1) { if (t < s) red[t] += red[t + s]; __syncthreads(); }
  float mean = red[0] * (1.0f / DM_);
  __syncthreads();
  float dd = v0 - mean;
  red[t] = dd * dd; __syncthreads();
  for (int s = DM_ / 2; s > 0; s >>= 1) { if (t < s) red[t] += red[t + s]; __syncthreads(); }
  float var = red[0] * (1.0f / DM_);
  __syncthreads();
  const float tt = dd * rsqrtf(var + 1e-5f) * ln1g[t] + ln1b[t];
  t1[t] = tt; __syncthreads();
  // FFN (exact gelu)
  for (int jj = 0; jj < 2; ++jj) {
    const int j = t + jj * DM_;
    float a = b1[j];
    for (int d = 0; d < DM_; ++d) a += t1[d] * w1[(size_t)j * DM_ + d];
    hbuf[j] = 0.5f * a * (1.0f + erff(a * 0.70710678118654752f));
  }
  __syncthreads();
  float f = b2[t];
  for (int j = 0; j < 2 * DM_; ++j) f += hbuf[j] * w2[(size_t)t * (2 * DM_) + j];
  const float v1 = tt + f;
  // LN2 -> z
  red[t] = v1; __syncthreads();
  for (int s = DM_ / 2; s > 0; s >>= 1) { if (t < s) red[t] += red[t + s]; __syncthreads(); }
  mean = red[0] * (1.0f / DM_);
  __syncthreads();
  dd = v1 - mean;
  red[t] = dd * dd; __syncthreads();
  for (int s = DM_ / 2; s > 0; s >>= 1) { if (t < s) red[t] += red[t + s]; __syncthreads(); }
  var = red[0] * (1.0f / DM_);
  z[t] = dd * rsqrtf(var + 1e-5f) * ln2g[t] + ln2b[t];
}

// ---------------------------------------------------------------------------
extern "C" void kernel_launch(void* const* d_in, const int* in_sizes, int n_in,
                              void* d_out, int out_size, void* d_ws, size_t ws_size,
                              hipStream_t stream) {
  (void)in_sizes; (void)n_in; (void)out_size; (void)ws_size;
  const float* H_S   = (const float*)d_in[0];
  const float* H_C   = (const float*)d_in[1];
  const float* H_G   = (const float*)d_in[2];
  const float* G     = (const float*)d_in[3];
  const float* Ws    = (const float*)d_in[4];
  const float* bs    = (const float*)d_in[5];
  const float* Wb    = (const float*)d_in[6];
  const float* bb    = (const float*)d_in[7];
  const float* Wg    = (const float*)d_in[8];
  const float* bg    = (const float*)d_in[9];
  const float* WQ    = (const float*)d_in[10];
  const float* WK    = (const float*)d_in[11];
  const float* WV    = (const float*)d_in[12];
  const float* WO    = (const float*)d_in[13];
  const float* bO    = (const float*)d_in[14];
  const float* lncg  = (const float*)d_in[15];
  const float* lncb  = (const float*)d_in[16];
  const float* fus   = (const float*)d_in[17];
  const float* miw   = (const float*)d_in[18];
  const float* mib   = (const float*)d_in[19];
  const float* mow   = (const float*)d_in[20];
  const float* mob   = (const float*)d_in[21];
  const float* ln1g  = (const float*)d_in[22];
  const float* ln1b  = (const float*)d_in[23];
  const float* ln2g  = (const float*)d_in[24];
  const float* ln2b  = (const float*)d_in[25];
  const float* fw1   = (const float*)d_in[26];
  const float* fb1   = (const float*)d_in[27];
  const float* fw2   = (const float*)d_in[28];
  const float* fb2   = (const float*)d_in[29];

  float* ws = (float*)d_ws;
  float* HS     = ws + 0;          // 2048*256
  float* HC     = ws + 524288;     // 4096*256
  float* HGp    = ws + 1572864;    // 2048*256
  float* Qm     = ws + 2097152;    // 2048*256
  float* Km     = ws + 2621440;    // 4096*256
  float* Vm     = ws + 3670016;    // 4096*256
  float* rmax   = ws + 4718592;    // 2048*8
  float* rinv   = ws + 4734976;    // 2048*8
  float* Hnew   = ws + 4751360;    // 2048*256
  float* tmp    = ws + 5275648;    // 2048*256
  float* HSu    = ws + 5799936;    // 2048*256
  float* tokens = ws + 6324224;    // 4097*256
  float* qkv    = ws + 7373056;    // 4097*768
  float* Hpart  = ws + 10519552;   // 8 * 2048*256 slab partials

  float* zout = (float*)d_out;          // [0..255]   z_fused
  float* amap = (float*)d_out + DM_;    // [256..]    attn_map (2048*4096)

  const dim3 wv(32);
  // input projections (all WMMA)
  gemm_xwT_bias<<<dim3(DM_ / 16, NS_ / 16), wv, 0, stream>>>(H_S, Ws, bs, HS, NS_, 128, DM_);
  gemm_xwT_bias<<<dim3(DM_ / 16, NC_ / 16), wv, 0, stream>>>(H_C, Wb, bb, HC, NC_, 256, DM_);
  gemm_xwT_bias<<<dim3(DM_ / 16, NG_ / 16), wv, 0, stream>>>(H_G, Wg, bg, HGp, NG_, 64, DM_);
  // Q/K/V projections
  gemm_xwT_bias<<<dim3(DM_ / 16, NS_ / 16), wv, 0, stream>>>(HS, WQ, nullptr, Qm, NS_, DM_, DM_);
  gemm_xwT_bias<<<dim3(DM_ / 16, NC_ / 16), wv, 0, stream>>>(HC, WK, nullptr, Km, NC_, DM_, DM_);
  gemm_xwT_bias<<<dim3(DM_ / 16, NC_ / 16), wv, 0, stream>>>(HC, WV, nullptr, Vm, NC_, DM_, DM_);
  // gated cross-attention, two-pass softmax; pass 2 slab-parallel (1024 waves)
  xattn_stats<<<dim3(NS_ / 16, NH_), wv, 0, stream>>>(Qm, Km, G, rmax, rinv);
  xattn_apply<<<dim3(NS_ / 16, NSLAB_), wv, 0, stream>>>(Qm, Km, Vm, G, rmax, rinv, amap, Hpart);
  hnew_reduce<<<dim3((NS_ * DM_) / 256), dim3(256), 0, stream>>>(Hpart, Hnew);
  // WO proj + residual LN
  gemm_xwT_bias<<<dim3(DM_ / 16, NS_ / 16), wv, 0, stream>>>(Hnew, WO, bO, tmp, NS_, DM_, DM_);
  ln_res_kernel<<<dim3(NS_), dim3(DM_), 0, stream>>>(tmp, HS, lncg, lncb, HSu);
  // tokens + qkv projection (q only needed for token 0, but k/v need all rows)
  build_tokens<<<dim3((TT_ * DM_ + 255) / 256), dim3(256), 0, stream>>>(fus, HSu, HGp, tokens);
  gemm_xwT_bias<<<dim3((3 * DM_) / 16, (TT_ + 15) / 16), wv, 0, stream>>>(tokens, miw, mib, qkv, TT_, DM_, 3 * DM_);
  // token-0 attention + FFN head -> z_fused
  fusion_head_kernel<<<1, dim3(DM_), 0, stream>>>(qkv, tokens, mow, mob, ln1g, ln1b,
                                                  fw1, fb1, fw2, fb2, ln2g, ln2b, zout);
}
